// Mask_Multi_Attn_52939766890895
// MI455X (gfx1250) — compile-verified, matrix operand order checked by reading, not run on hardware
//
#include <hip/hip_runtime.h>
#include <hip/hip_bf16.h>

typedef __attribute__((ext_vector_type(16))) _Float16 v16h;
typedef __attribute__((ext_vector_type(8)))  _Float16 v8h;
typedef __attribute__((ext_vector_type(8)))  float    v8f;

#define HEADS  16
#define DMODEL 1024
#define KDIM   64
#define SEGN   1024
#define MEMN   2048
#define KTOT   3072

// ---------------------------------------------------------------------------
// WMMA helpers (CDNA5 wave32, v_wmma_f32_16x16x32_f16)
// ---------------------------------------------------------------------------
__device__ __forceinline__ v8f wmma16(v16h a, v16h b, v8f c) {
  return __builtin_amdgcn_wmma_f32_16x16x32_f16(false, a, false, b, (short)0, c,
                                                false, false);
}

// Load a 16x32 f16 A-fragment (or 32x16 B-fragment from a K-contiguous /
// "transposed" matrix). Per ISA layout: lane = half*16 + row, VGPR i holds
// k = (i>>2)*16 + half*8 + (i&3)*2 -> two contiguous 16B chunks per lane.
__device__ __forceinline__ v16h load_frag(const _Float16* __restrict__ p,
                                          int stride, int row, int kbase,
                                          int lane) {
  const int lm = lane & 15, hf = lane >> 4;
  const _Float16* q = p + (size_t)(row + lm) * stride + (kbase + hf * 8);
  v8h lo = *(const v8h*)(q);
  v8h hi = *(const v8h*)(q + 16);
  return __builtin_shufflevector(lo, hi, 0, 1, 2, 3, 4, 5, 6, 7, 8, 9, 10, 11,
                                 12, 13, 14, 15);
}

__device__ __forceinline__ v16h load_frag_clamp(const _Float16* __restrict__ p,
                                                int stride, int row, int maxrow,
                                                int kbase, int lane) {
  const int lm = lane & 15, hf = lane >> 4;
  int r = row + lm;
  if (r > maxrow) r = maxrow;
  const _Float16* q = p + (size_t)r * stride + (kbase + hf * 8);
  v8h lo = *(const v8h*)(q);
  v8h hi = *(const v8h*)(q + 16);
  return __builtin_shufflevector(lo, hi, 0, 1, 2, 3, 4, 5, 6, 7, 8, 9, 10, 11,
                                 12, 13, 14, 15);
}

// Async global->LDS 16B copy (CDNA5 Tensor/async path, ASYNCcnt-tracked).
// lds = workgroup-relative LDS byte offset (low 32 bits of the flat address
// of a __shared__ object per ISA 10.2 aperture mapping).
__device__ __forceinline__ void async_cp16(unsigned lds, const _Float16* g) {
  asm volatile("global_load_async_to_lds_b128 %0, %1, off"
               :
               : "v"(lds), "v"((unsigned long long)(uintptr_t)g)
               : "memory");
}

// ---------------------------------------------------------------------------
// Prep kernels
// ---------------------------------------------------------------------------
__global__ void k_cvt(const float* __restrict__ in, _Float16* __restrict__ out,
                      int n) {
  int i = blockIdx.x * blockDim.x + threadIdx.x;
  int stride = gridDim.x * blockDim.x;
  for (; i < n; i += stride) out[i] = (_Float16)in[i];
}

// transpose 1024x1024 f32 -> f16 (out[n][k] = in[k][n]) via LDS tiles
__global__ void k_cvtT(const float* __restrict__ in,
                       _Float16* __restrict__ out) {
  __shared__ _Float16 tile[32][33];
  const int bx = blockIdx.x * 32;  // input col / output row block
  const int by = blockIdx.y * 32;  // input row / output col block
  for (int i = threadIdx.y; i < 32; i += 8)
    tile[i][threadIdx.x] = (_Float16)in[(size_t)(by + i) * DMODEL + bx + threadIdx.x];
  __syncthreads();
  for (int i = threadIdx.y; i < 32; i += 8)
    out[(size_t)(bx + i) * DMODEL + by + threadIdx.x] = tile[threadIdx.x][i];
}

// fused biases: bq + tile(u) , bq + tile(v)
__global__ void k_bias(const float* __restrict__ bq, const float* __restrict__ u,
                       const float* __restrict__ v, float* __restrict__ bqu,
                       float* __restrict__ bqv) {
  int d = blockIdx.x * blockDim.x + threadIdx.x;
  if (d < DMODEL) {
    bqu[d] = bq[d] + u[d & 63];
    bqv[d] = bq[d] + v[d & 63];
  }
}

// rel = R @ Wp + bp  (3072x64, tiny GEMM) -> f16 row-major
__global__ void k_rel(const float* __restrict__ R, const float* __restrict__ Wp,
                      const float* __restrict__ bp,
                      _Float16* __restrict__ relH) {
  int idx = blockIdx.x * blockDim.x + threadIdx.x;
  if (idx >= KTOT * KDIM) return;
  int s = idx >> 6, d = idx & 63;
  float acc = bp[d];
  for (int k = 0; k < KDIM; ++k) acc += R[s * KDIM + k] * Wp[k * KDIM + d];
  relH[idx] = (_Float16)acc;
}

// ---------------------------------------------------------------------------
// WMMA GEMM: C[MxN] = A[Mx1024] @ BT^T + bias, N = 1024 fixed.
// BT is the weight stored transposed (N x 1024) so B-fragments load
// K-contiguously. 4 waves/block, each wave computes a 32x32 tile.
//
// Operand staging uses the CDNA5 async path: a double-buffered
// global_load_async_to_lds_b128 pipeline (2 K-steps in flight, ASYNCcnt
// retires in order -> s_wait_asynccnt 8 frees the oldest batch) so latency is
// hidden without holding 64 in-flight destination VGPRs. Each lane reads back
// only its own LDS slots (ds_load_b128), so no cross-wave barrier is needed;
// s_wait_dscnt 0 guards the WAR before a buffer is re-issued (async LDS
// writes are unordered vs DS ops).
//
// mode 0: f16 row-major out; mode 1: f32 row-major out;
// mode 2: f16 transposed out (outT[n*M + m]) with packed b128 stores.
// ---------------------------------------------------------------------------
__global__ void __launch_bounds__(128) k_gemm(const _Float16* __restrict__ A,
                                              const _Float16* __restrict__ BT,
                                              const float* __restrict__ bias,
                                              void* __restrict__ out, int M,
                                              int mode) {
  // [buf][wave][chunk(8)][lane(32)][8 halfs] : 2*4*8*32*16B = 32 KB
  __shared__ __align__(16) _Float16 stage[2 * 4 * 8 * 32 * 8];

  const int lane = threadIdx.x & 31;
  const int wave = threadIdx.x >> 5;
  const int lm = lane & 15, hf = lane >> 4;
  const int m0 = blockIdx.y * 64 + (wave >> 1) * 32;
  const int n0 = blockIdx.x * 64 + (wave & 1) * 32;

  const unsigned ldsbase =
      (unsigned)(uintptr_t)(void*)stage +
      (unsigned)((wave * 8 * 32 + lane) * 8 * sizeof(_Float16));
  const unsigned bufstride = 4u * 8u * 32u * 8u * sizeof(_Float16);  // 16 KB
  const unsigned chkstride = 32u * 8u * sizeof(_Float16);            // 512 B

  const _Float16* pa0 = A + (size_t)(m0 + lm) * DMODEL + hf * 8;
  const _Float16* pa1 = A + (size_t)(m0 + 16 + lm) * DMODEL + hf * 8;
  const _Float16* pb0 = BT + (size_t)(n0 + lm) * DMODEL + hf * 8;
  const _Float16* pb1 = BT + (size_t)(n0 + 16 + lm) * DMODEL + hf * 8;

  auto issue = [&](int kk, int buf) {
    const unsigned b = ldsbase + (unsigned)buf * bufstride;
    async_cp16(b + 0 * chkstride, pa0 + kk);
    async_cp16(b + 1 * chkstride, pa0 + kk + 16);
    async_cp16(b + 2 * chkstride, pa1 + kk);
    async_cp16(b + 3 * chkstride, pa1 + kk + 16);
    async_cp16(b + 4 * chkstride, pb0 + kk);
    async_cp16(b + 5 * chkstride, pb0 + kk + 16);
    async_cp16(b + 6 * chkstride, pb1 + kk);
    async_cp16(b + 7 * chkstride, pb1 + kk + 16);
  };
  const _Float16* mystage = stage + (size_t)(wave * 8 * 32 + lane) * 8;
  auto frag = [&](int buf, int c) -> v16h {
    const _Float16* p = mystage + (size_t)buf * (4 * 8 * 32 * 8) +
                        (size_t)(2 * c) * (32 * 8);
    v8h lo = *(const v8h*)p;
    v8h hi = *(const v8h*)(p + 32 * 8);
    return __builtin_shufflevector(lo, hi, 0, 1, 2, 3, 4, 5, 6, 7, 8, 9, 10,
                                   11, 12, 13, 14, 15);
  };

  issue(0, 0);
  issue(32, 1);
  int buf = 0;
  v8f acc[2][2] = {};
  for (int k0 = 0; k0 < DMODEL; k0 += 32) {
    if (k0 + 32 < DMODEL)
      asm volatile("s_wait_asynccnt 0x8" ::: "memory");  // oldest batch done
    else
      asm volatile("s_wait_asynccnt 0x0" ::: "memory");  // drain
    v16h a0 = frag(buf, 0);
    v16h a1 = frag(buf, 1);
    v16h b0 = frag(buf, 2);
    v16h b1 = frag(buf, 3);
    acc[0][0] = wmma16(a0, b0, acc[0][0]);
    acc[0][1] = wmma16(a0, b1, acc[0][1]);
    acc[1][0] = wmma16(a1, b0, acc[1][0]);
    acc[1][1] = wmma16(a1, b1, acc[1][1]);
    if (k0 + 64 < DMODEL) {
      asm volatile("s_wait_dscnt 0x0" ::: "memory");  // WAR on this buffer
      issue(k0 + 64, buf);
    }
    buf ^= 1;
  }

  for (int ti = 0; ti < 2; ++ti) {
    for (int tj = 0; tj < 2; ++tj) {
      const int n = n0 + tj * 16 + lm;
      const float bb = bias[n];
      const int mb = m0 + ti * 16 + hf * 8;  // D-frag: m = mb + r
      if (mode == 0) {
        _Float16* o = (_Float16*)out;
#pragma unroll
        for (int r = 0; r < 8; ++r)
          o[(size_t)(mb + r) * DMODEL + n] = (_Float16)(acc[ti][tj][r] + bb);
      } else if (mode == 1) {
        float* o = (float*)out;
#pragma unroll
        for (int r = 0; r < 8; ++r)
          o[(size_t)(mb + r) * DMODEL + n] = acc[ti][tj][r] + bb;
      } else {
        _Float16* o = (_Float16*)out;
        v8h pk;
#pragma unroll
        for (int r = 0; r < 8; ++r) pk[r] = (_Float16)(acc[ti][tj][r] + bb);
        *(v8h*)(o + (size_t)n * M + mb) = pk;  // 16B packed store
      }
    }
  }
}

// ---------------------------------------------------------------------------
// Fused relative attention for one (head, 16-row tile).
// 4 waves; scores (16 x 3072) kept as f16 in dynamic LDS.
// bd shift: score[m][jt+n] needs bd_[m][base + 15 + n - m],
// base = jt + SEG - 16 - i0 (16-aligned) -> exactly two aligned bd_ tiles.
// ---------------------------------------------------------------------------
__global__ void __launch_bounds__(128) k_attn(
    const _Float16* __restrict__ quH, const _Float16* __restrict__ qvH,
    const _Float16* __restrict__ kH, const _Float16* __restrict__ relH,
    const _Float16* __restrict__ vTH, _Float16* __restrict__ attnH) {
  extern __shared__ char smem[];
  const int SCS = KTOT + 8;                      // padded row stride (halfs)
  _Float16* sc = (_Float16*)smem;                // [16][SCS] f16 scores
  float* bdst = (float*)(smem + 16 * SCS * 2);   // [4][16][32] bd staging
  float* red = bdst + 4 * 16 * 32;               // [16][8]
  float* rowMax = red + 128;                     // [16]
  float* rowInv = rowMax + 16;                   // [16]

  const int h = blockIdx.x;
  const int i0 = blockIdx.y * 16;
  const int lane = threadIdx.x & 31;
  const int wave = threadIdx.x >> 5;
  const int lm = lane & 15, hf = lane >> 4;

  // ---- Phase 1: scores = (q+u)K^T + shift((q+v)rel^T), mask -> f16 LDS ----
  v16h aqu0 = load_frag(quH, DMODEL, i0, h * 64, lane);
  v16h aqu1 = load_frag(quH, DMODEL, i0, h * 64 + 32, lane);
  v16h aqv0 = load_frag(qvH, DMODEL, i0, h * 64, lane);
  v16h aqv1 = load_frag(qvH, DMODEL, i0, h * 64 + 32, lane);
  float* mybd = bdst + wave * 512;

  for (int t = 0; t < 48; ++t) {
    const int jt = (wave * 48 + t) * 16;
    v8f ac = {};
    v16h b0 = load_frag(kH, DMODEL, jt, h * 64, lane);
    v16h b1 = load_frag(kH, DMODEL, jt, h * 64 + 32, lane);
    ac = wmma16(aqu0, b0, ac);
    ac = wmma16(aqu1, b1, ac);

    const int base = jt + SEGN - 16 - i0;  // >= 0, 16-aligned
#pragma unroll
    for (int tt = 0; tt < 2; ++tt) {
      v8f bd = {};
      v16h r0 = load_frag_clamp(relH, KDIM, base + tt * 16, KTOT - 1, 0, lane);
      v16h r1 = load_frag_clamp(relH, KDIM, base + tt * 16, KTOT - 1, 32, lane);
      bd = wmma16(aqv0, r0, bd);
      bd = wmma16(aqv1, r1, bd);
#pragma unroll
      for (int r = 0; r < 8; ++r)
        mybd[(r + 8 * hf) * 32 + tt * 16 + lm] = bd[r];
    }
    asm volatile("s_wait_dscnt 0" ::: "memory");  // cross-lane LDS RAW
#pragma unroll
    for (int r = 0; r < 8; ++r) {
      const int m = r + 8 * hf;
      const int n = lm;
      float s = ac[r] + mybd[m * 32 + 15 + n - m];
      if (jt + n > MEMN + i0 + m) s = -60000.0f;  // causal mask
      sc[m * SCS + jt + n] = (_Float16)s;
    }
  }
  __syncthreads();

  // ---- Phase 2: per-row softmax stats (max, 1/sum) ----
  {
    const int r = threadIdx.x >> 3, sub = threadIdx.x & 7;
    float mx = -1e30f;
    for (int c = sub; c < KTOT; c += 8) mx = fmaxf(mx, (float)sc[r * SCS + c]);
    red[r * 8 + sub] = mx;
    __syncthreads();
    if (sub == 0) {
      float m2 = red[r * 8];
      for (int q = 1; q < 8; ++q) m2 = fmaxf(m2, red[r * 8 + q]);
      rowMax[r] = m2;
    }
    __syncthreads();
    const float Mx = rowMax[r];
    float sum = 0.f;
    for (int c = sub; c < KTOT; c += 8) sum += __expf((float)sc[r * SCS + c] - Mx);
    __syncthreads();
    red[r * 8 + sub] = sum;
    __syncthreads();
    if (sub == 0) {
      float s2 = 0.f;
      for (int q = 0; q < 8; ++q) s2 += red[r * 8 + q];
      rowInv[r] = 1.0f / s2;
    }
    __syncthreads();
  }

  // ---- Phase 3: out = softmax(scores) @ V  (one 16-col tile per wave) ----
  {
    const float mym = rowMax[lm];
    const float myi = rowInv[lm];
    const int n0 = h * 64 + wave * 16;  // column base in attnH / row base in vT
    v8f acc = {};
    for (int k0 = 0; k0 < KTOT; k0 += 32) {
      const _Float16* sp = sc + lm * SCS + k0 + hf * 8;
      v8h slo = *(const v8h*)sp;
      v8h shi = *(const v8h*)(sp + 16);
      v16h pa;
#pragma unroll
      for (int e = 0; e < 8; ++e)
        pa[e] = (_Float16)(__expf((float)slo[e] - mym) * myi);
#pragma unroll
      for (int e = 0; e < 8; ++e)
        pa[8 + e] = (_Float16)(__expf((float)shi[e] - mym) * myi);
      v16h vb = load_frag(vTH, KTOT, n0, k0, lane);
      acc = wmma16(pa, vb, acc);
    }
#pragma unroll
    for (int r = 0; r < 8; ++r)
      attnH[(size_t)(i0 + r + 8 * hf) * DMODEL + n0 + lm] = (_Float16)acc[r];
  }
}

// ---------------------------------------------------------------------------
// Host launcher
// ---------------------------------------------------------------------------
extern "C" void kernel_launch(void* const* d_in, const int* in_sizes, int n_in,
                              void* d_out, int out_size, void* d_ws,
                              size_t ws_size, hipStream_t stream) {
  (void)in_sizes; (void)n_in; (void)out_size; (void)ws_size;
  const float* last_src = (const float*)d_in[0];
  const float* src = (const float*)d_in[1];
  const float* R = (const float*)d_in[2];
  const float* u_vec = (const float*)d_in[3];
  const float* v_vec = (const float*)d_in[4];
  const float* Wq = (const float*)d_in[5];
  const float* bq = (const float*)d_in[6];
  const float* Wk = (const float*)d_in[7];
  const float* bk = (const float*)d_in[8];
  const float* Wp = (const float*)d_in[9];
  const float* bp = (const float*)d_in[10];
  const float* Wv = (const float*)d_in[11];
  const float* bv = (const float*)d_in[12];
  const float* Wo = (const float*)d_in[13];
  const float* bo = (const float*)d_in[14];
  // d_in[15] = mask: computed analytically, unused.

  char* ws = (char*)d_ws;
  size_t off = 0;
  auto carve = [&](size_t bytes) -> char* {
    char* p = ws + off;
    off += (bytes + 255) & ~(size_t)255;
    return p;
  };
  _Float16* srcH = (_Float16*)carve(sizeof(_Float16) * SEGN * DMODEL);
  _Float16* lstateH = (_Float16*)carve(sizeof(_Float16) * KTOT * DMODEL);
  _Float16* WqT = (_Float16*)carve(sizeof(_Float16) * DMODEL * DMODEL);
  _Float16* WkT = (_Float16*)carve(sizeof(_Float16) * DMODEL * DMODEL);
  _Float16* WvT = (_Float16*)carve(sizeof(_Float16) * DMODEL * DMODEL);
  _Float16* WoT = (_Float16*)carve(sizeof(_Float16) * DMODEL * DMODEL);
  _Float16* quH = (_Float16*)carve(sizeof(_Float16) * SEGN * DMODEL);
  _Float16* qvH = (_Float16*)carve(sizeof(_Float16) * SEGN * DMODEL);
  _Float16* kH = (_Float16*)carve(sizeof(_Float16) * KTOT * DMODEL);
  _Float16* vTH = (_Float16*)carve(sizeof(_Float16) * DMODEL * KTOT);
  _Float16* relH = (_Float16*)carve(sizeof(_Float16) * KTOT * KDIM);
  _Float16* attnH = (_Float16*)carve(sizeof(_Float16) * SEGN * DMODEL);
  float* biasQU = (float*)carve(sizeof(float) * DMODEL);
  float* biasQV = (float*)carve(sizeof(float) * DMODEL);

  // 1) conversions
  k_cvt<<<1024, 256, 0, stream>>>(src, srcH, SEGN * DMODEL);
  k_cvt<<<2048, 256, 0, stream>>>(last_src, lstateH, MEMN * DMODEL);
  k_cvt<<<1024, 256, 0, stream>>>(src, lstateH + (size_t)MEMN * DMODEL,
                                  SEGN * DMODEL);
  k_cvtT<<<dim3(32, 32), dim3(32, 8), 0, stream>>>(Wq, WqT);
  k_cvtT<<<dim3(32, 32), dim3(32, 8), 0, stream>>>(Wk, WkT);
  k_cvtT<<<dim3(32, 32), dim3(32, 8), 0, stream>>>(Wv, WvT);
  k_cvtT<<<dim3(32, 32), dim3(32, 8), 0, stream>>>(Wo, WoT);
  k_bias<<<4, 256, 0, stream>>>(bq, u_vec, v_vec, biasQU, biasQV);
  k_rel<<<(KTOT * KDIM + 255) / 256, 256, 0, stream>>>(R, Wp, bp, relH);

  // 2) projections (WMMA + async-to-LDS pipeline)
  k_gemm<<<dim3(16, 16), 128, 0, stream>>>(srcH, WqT, biasQU, quH, SEGN, 0);
  k_gemm<<<dim3(16, 16), 128, 0, stream>>>(srcH, WqT, biasQV, qvH, SEGN, 0);
  k_gemm<<<dim3(16, 48), 128, 0, stream>>>(lstateH, WkT, bk, kH, KTOT, 0);
  k_gemm<<<dim3(16, 48), 128, 0, stream>>>(lstateH, WvT, bv, vTH, KTOT, 2);

  // 3) fused relative attention (WMMA + softmax)
  const int smem = 16 * (KTOT + 8) * 2 + 4 * 16 * 32 * 4 + 128 * 4 + 32 * 4;
  k_attn<<<dim3(HEADS, SEGN / 16), 128, smem, stream>>>(quH, qvH, kH, relH,
                                                        vTH, attnH);

  // 4) output projection -> f32 d_out
  k_gemm<<<dim3(16, 16), 128, 0, stream>>>(attnH, WoT, bo, d_out, SEGN, 1);
}
